// TATLMPLayer_44925357916948
// MI455X (gfx1250) — matrix operand order repaired and checked by reading, not compiled, not run on hardware
//
#include <hip/hip_runtime.h>
#include <hip/hip_fp16.h>
#include <stdint.h>
#include <stddef.h>

// Problem constants (from setup_inputs)
#define BB 2
#define NN 256
#define EE 512
#define PP 256
#define DD 128
#define TT 8
#define KK 8
#define HD 256

typedef __attribute__((ext_vector_type(16))) _Float16 v16h;
typedef __attribute__((ext_vector_type(8)))  float    v8f;

union Frag { v16h v; uint32_t u[8]; };

// ---------------------------------------------------------------------------
// Generic single-wave WMMA GEMM: C[b] = act( scale * A[b] @ B[b] + bias )
// A: f16 row-major MxK (lda). B is ALWAYS given transposed: element (k,n) at
// Bm[n*ldb+k]  (i.e. Bm is N x K row-major) -> two contiguous 16B loads per fragment.
// C: f32 or f16 (OUTF16), optionally stored transposed (CT: C[n*ldc+m]).
// One wave computes a 16x16 C tile. M,N mult of 16; K mult of 32.
// ---------------------------------------------------------------------------
struct GemmP {
  const __half* A;  long long sA; int lda;
  const __half* Bm; long long sB; int ldb;
  void* C;          long long sC; int ldc;
  const float* bias; float scale;
  int Kd;
};

template <int OUTF16, int RELU, int CT>
__global__ __launch_bounds__(32) void gemm16(GemmP p) {
  const int lane = threadIdx.x;
  const int tn = blockIdx.x, tm = blockIdx.y, bz = blockIdx.z;
  const __half* A  = p.A  + (long long)bz * p.sA;
  const __half* Bm = p.Bm + (long long)bz * p.sB;
  const int l15 = lane & 15;
  const int kh  = lane >> 4;
  const int m = tm * 16 + l15;
  const int n = tn * 16 + l15;
  v8f c = {};
  for (int kb = 0; kb < p.Kd; kb += 32) {
    Frag a, b;
    // A fragment: 16-bit A 16x32 layout. dwords [4*kh .. 4*kh+3] then
    // [8+4*kh .. 8+4*kh+3] -> two 16B contiguous groups.
    const uint32_t* ap = (const uint32_t*)(A + (size_t)m * p.lda + kb);
#pragma unroll
    for (int i = 0; i < 8; ++i) {
      int k0 = (i < 4) ? (8 * kh + 2 * i) : (16 + 8 * kh + 2 * (i - 4));
      a.u[i] = ap[k0 >> 1];
    }
    // B fragment (transposed storage): same k order as A at col n.
    const uint32_t* bp = (const uint32_t*)(Bm + (size_t)n * p.ldb + kb);
#pragma unroll
    for (int i = 0; i < 8; ++i) {
      int k0 = (i < 4) ? (8 * kh + 2 * i) : (16 + 8 * kh + 2 * (i - 4));
      b.u[i] = bp[k0 >> 1];
    }
    c = __builtin_amdgcn_wmma_f32_16x16x32_f16(false, a.v, false, b.v,
                                               (short)0, c, false, false);
  }
  const float bsv = p.bias ? p.bias[n] : 0.0f;
#pragma unroll
  for (int r = 0; r < 8; ++r) {
    int row = tm * 16 + r + 8 * kh;
    float v = c[r] * p.scale + bsv;
    if (RELU) v = fmaxf(v, 0.0f);
    size_t ci = CT ? ((size_t)n * p.ldc + row) : ((size_t)row * p.ldc + n);
    if (OUTF16) {
      __half* Cp = (__half*)p.C + (long long)bz * p.sC;
      Cp[ci] = __float2half(v);
    } else {
      float* Cp = (float*)p.C + (long long)bz * p.sC;
      Cp[ci] = v;
    }
  }
}

// ---------------------------------------------------------------------------
// Elementwise / helper kernels
// ---------------------------------------------------------------------------
__global__ void k_cvt(const float* s, __half* d, int n) {
  int i = blockIdx.x * 256 + threadIdx.x;
  if (i < n) d[i] = __float2half(s[i]);
}

// convert + transpose: src (bat, rows, cols) f32 -> dst (bat, cols, rows) f16
__global__ void k_cvtT(const float* s, __half* d, int bat, int rows, int cols) {
  int i = blockIdx.x * 256 + threadIdx.x;
  int tot = bat * rows * cols;
  if (i >= tot) return;
  int b = i / (rows * cols), rem = i - b * rows * cols;
  int r = rem / cols, c = rem - r * cols;
  d[(size_t)b * rows * cols + (size_t)c * rows + r] = __float2half(s[i]);
}

// pad rows to dstR then transpose: dst[c*dstR + r], zero for r >= srcR
__global__ void k_padrowsT(const float* s, __half* d, int srcR, int dstR, int cols) {
  int i = blockIdx.x * 256 + threadIdx.x;
  int tot = dstR * cols;
  if (i >= tot) return;
  int r = i / cols, c = i - r * cols;
  d[(size_t)c * dstR + r] = __float2half(r < srcR ? s[r * cols + c] : 0.0f);
}

// pad cols to dstC then transpose: dst[c*rows + r], zero for c >= srcC
__global__ void k_padcolsT(const float* s, __half* d, int rows, int srcC, int dstC) {
  int i = blockIdx.x * 256 + threadIdx.x;
  int tot = rows * dstC;
  if (i >= tot) return;
  int r = i / dstC, c = i - r * dstC;
  d[(size_t)c * rows + r] = __float2half(c < srcC ? s[r * srcC + c] : 0.0f);
}

__global__ void k_padbias(const float* s, float* d, int srcN, int dstN) {
  int i = blockIdx.x * 256 + threadIdx.x;
  if (i < dstN) d[i] = (i < srcN) ? s[i] : 0.0f;
}

// c0 = relu(w_b1) @ w_W2 + w_b2   (MLP output for an all-zero input)
__global__ __launch_bounds__(256) void k_c0(const float* b1, const float* W2,
                                            const float* b2, float* c0) {
  __shared__ float red[256];
  int t = threadIdx.x;
  red[t] = fmaxf(b1[t], 0.0f) * W2[t];
  __syncthreads();
  for (int s = 128; s > 0; s >>= 1) {
    if (t < s) red[t] += red[t + s];
    __syncthreads();
  }
  if (t == 0) c0[0] = red[0] + b2[0];
}

// Level-1: ew = relu(U[e]+V[n])@W2 + b2 ; Hw = softmax(ew*H, over n) * H
__global__ __launch_bounds__(256) void k_pair1(const float* U, const float* V,
                                               const float* W2, const float* b2,
                                               const float* H, __half* Hw) {
  int e = blockIdx.x, b = blockIdx.y, t = threadIdx.x;
  __shared__ float uS[HD], wS[HD], red[256];
  long idx = (long)(b * EE + e);
  uS[t] = U[idx * HD + t];
  wS[t] = W2[t];
  __syncthreads();
  const float* vr = V + ((long)b * NN + t) * HD;
  float acc = 0.0f;
  for (int h = 0; h < HD; ++h) acc += fmaxf(uS[h] + vr[h], 0.0f) * wS[h];
  float ew = acc + b2[0];
  float hm = H[idx * NN + t];
  float x = ew * hm;
  red[t] = x; __syncthreads();
  for (int s = 128; s > 0; s >>= 1) { if (t < s) red[t] = fmaxf(red[t], red[t + s]); __syncthreads(); }
  float mx = red[0]; __syncthreads();
  float p = __expf(x - mx);
  red[t] = p; __syncthreads();
  for (int s = 128; s > 0; s >>= 1) { if (t < s) red[t] += red[t + s]; __syncthreads(); }
  float sm = red[0];
  Hw[idx * NN + t] = __float2half(p / sm * hm);
}

// Level-2: raw = H ? relu(U2[e]+V[n])@W2+b2 : c0 ; leaky-relu; softmax over n;
// Hw2 = p/sum * H ; stored TRANSPOSED (B,N,E) for the msg GEMM.
__global__ __launch_bounds__(256) void k_pair2(const float* U2, const float* V,
                                               const float* W2, const float* b2,
                                               const float* H, const float* c0,
                                               __half* Hw2T) {
  int e = blockIdx.x, b = blockIdx.y, t = threadIdx.x;
  __shared__ float uS[HD], wS[HD], red[256];
  long idx = (long)(b * EE + e);
  uS[t] = U2[idx * HD + t];
  wS[t] = W2[t];
  __syncthreads();
  const float* vr = V + ((long)b * NN + t) * HD;
  float acc = 0.0f;
  for (int h = 0; h < HD; ++h) acc += fmaxf(uS[h] + vr[h], 0.0f) * wS[h];
  float hm = H[idx * NN + t];
  float x = (hm != 0.0f) ? (acc + b2[0]) : c0[0];
  x = (x > 0.0f) ? x : 0.01f * x;  // leaky relu
  red[t] = x; __syncthreads();
  for (int s = 128; s > 0; s >>= 1) { if (t < s) red[t] = fmaxf(red[t], red[t + s]); __syncthreads(); }
  float mx = red[0]; __syncthreads();
  float p = __expf(x - mx);
  red[t] = p; __syncthreads();
  for (int s = 128; s > 0; s >>= 1) { if (t < s) red[t] += red[t + s]; __syncthreads(); }
  float sm = red[0];
  Hw2T[((long)b * NN + t) * EE + e] = __float2half(p / sm * hm);
}

// edge_type = H @ agent_types, written f16 into etpad cols 128..159 (pad 136..159 zero)
__global__ __launch_bounds__(32) void k_etype(const float* H, const float* at, __half* etpad) {
  int e = blockIdx.x, b = blockIdx.y, t = threadIdx.x;
  long idx = (long)(b * EE + e);
  float acc = 0.0f;
  if (t < TT) {
    const float* hr = H + idx * NN;
    for (int n2 = 0; n2 < NN; ++n2) acc += hr[n2] * at[((long)b * NN + n2) * TT + t];
  }
  etpad[idx * 160 + 128 + t] = __float2half(t < TT ? acc : 0.0f);
}

__device__ inline float urnd(uint32_t x) {
  x ^= x >> 17; x *= 0xed5ad4bbu; x ^= x >> 11; x *= 0xac4c1b51u;
  x ^= x >> 15; x *= 0x31848babu; x ^= x >> 14;
  float u = (float)(x >> 8) * (1.0f / 16777216.0f);
  return fminf(fmaxf(u, 1e-6f), 1.0f - 1e-6f);
}

// gumbel-softmax typed embedding: ete = (sigmoid(fraw)*softmax((logits+g)*2)) @ type_emb
// also assembles EF2^T (256 x E per batch) = [edge_features | ete]^T
__global__ __launch_bounds__(128) void k_ete(const float* logits, const float* fraw,
                                             const float* temb, const __half* etpad,
                                             __half* ete, __half* EF2T) {
  int e = blockIdx.x, b = blockIdx.y, d = threadIdx.x;
  long idx = (long)(b * EE + e);
  float z[KK];
  float mx = -1e30f;
#pragma unroll
  for (int k = 0; k < KK; ++k) {
    float g = -__logf(-__logf(urnd((uint32_t)(idx * KK + k) * 2654435761u + 12345u)));
    z[k] = (logits[idx * 16 + k] + g) * 2.0f;  // / tau(=0.5)
    mx = fmaxf(mx, z[k]);
  }
  float s = 0.0f;
#pragma unroll
  for (int k = 0; k < KK; ++k) { z[k] = __expf(z[k] - mx); s += z[k]; }
  float fac = 1.0f / (1.0f + __expf(-fraw[idx * 16]));
  float acc = 0.0f;
#pragma unroll
  for (int k = 0; k < KK; ++k) acc += (z[k] / s) * temb[k * DD + d];
  __half h = __float2half(fac * acc);
  ete[idx * DD + d] = h;
  __half* ebase = EF2T + (long)b * 256 * EE;
  ebase[(long)(128 + d) * EE + e] = h;
  ebase[(long)d * EE + e] = etpad[idx * 160 + d];
}

// cat2 cols 0..383 = [msg | agent_features] (f16)
__global__ void k_cat2a(const float* msg, const float* af, __half* cat2) {
  int i = blockIdx.x * 256 + threadIdx.x;
  int tot = BB * NN * 384;
  if (i >= tot) return;
  int bn = i / 384, c = i - bn * 384;
  float v = (c < 256) ? msg[(long)bn * 256 + c] : af[(long)bn * DD + (c - 256)];
  cat2[(long)bn * 512 + c] = __float2half(v);
}

// cat2 cols 384..511 = polyline_new (f16 from f32)
__global__ void k_cat2b(const float* pn, __half* cat2) {
  int i = blockIdx.x * 256 + threadIdx.x;
  int tot = BB * NN * DD;
  if (i >= tot) return;
  int bn = i / DD, c = i - bn * DD;
  cat2[(long)bn * 512 + 384 + c] = __float2half(pn[i]);
}

// row softmax of attention scores (B,4,N,P) -> f16
__global__ __launch_bounds__(256) void k_softrow(const float* sc, __half* att) {
  __shared__ float red[256];
  long r = blockIdx.x;
  int t = threadIdx.x;
  float x = sc[r * PP + t];
  red[t] = x; __syncthreads();
  for (int s = 128; s > 0; s >>= 1) { if (t < s) red[t] = fmaxf(red[t], red[t + s]); __syncthreads(); }
  float mx = red[0]; __syncthreads();
  float p = __expf(x - mx);
  red[t] = p; __syncthreads();
  for (int s = 128; s > 0; s >>= 1) { if (t < s) red[t] += red[t + s]; __syncthreads(); }
  att[r * PP + t] = __float2half(p / red[0]);
}

// ---------------------------------------------------------------------------
extern "C" void kernel_launch(void* const* d_in, const int* in_sizes, int n_in,
                              void* d_out, int out_size, void* d_ws, size_t ws_size,
                              hipStream_t stream) {
  (void)in_sizes; (void)n_in; (void)out_size; (void)ws_size;
  const float* af  = (const float*)d_in[0];
  const float* at  = (const float*)d_in[1];
  const float* pf  = (const float*)d_in[2];
  const float* H   = (const float*)d_in[3];
  const float* temb= (const float*)d_in[4];
  const float* wW1 = (const float*)d_in[5];  const float* wb1 = (const float*)d_in[6];
  const float* wW2 = (const float*)d_in[7];  const float* wb2 = (const float*)d_in[8];
  const float* dW1 = (const float*)d_in[9];  const float* db1 = (const float*)d_in[10];
  const float* dW2 = (const float*)d_in[11]; const float* db2 = (const float*)d_in[12];
  const float* fW1 = (const float*)d_in[13]; const float* fb1 = (const float*)d_in[14];
  const float* fW2 = (const float*)d_in[15]; const float* fb2 = (const float*)d_in[16];
  const float* oW1 = (const float*)d_in[17]; const float* ob1 = (const float*)d_in[18];
  const float* oW2 = (const float*)d_in[19]; const float* ob2 = (const float*)d_in[20];
  const float* qW  = (const float*)d_in[21]; const float* qb  = (const float*)d_in[22];
  const float* kW  = (const float*)d_in[23]; const float* kb  = (const float*)d_in[24];
  const float* vW  = (const float*)d_in[25]; const float* vb  = (const float*)d_in[26];
  const float* iqW = (const float*)d_in[27]; const float* iqb = (const float*)d_in[28];
  const float* ikW = (const float*)d_in[29]; const float* ikb = (const float*)d_in[30];
  const float* ivW = (const float*)d_in[31]; const float* ivb = (const float*)d_in[32];
  const float* opW = (const float*)d_in[33]; const float* opb = (const float*)d_in[34];

  size_t off = 0;
  auto alloc = [&](size_t bytes) -> void* {
    size_t o = (off + 255) & ~(size_t)255;
    off = o + bytes;
    return (void*)((char*)d_ws + o);
  };
  // f16 buffers (A operands row-major; all B operands stored transposed N x K)
  __half* af16  = (__half*)alloc((size_t)BB*NN*DD*2);   // A: (B,N,D)
  __half* af16T = (__half*)alloc((size_t)BB*DD*NN*2);   // B^T: (B,D,N)
  __half* pf16  = (__half*)alloc((size_t)BB*PP*DD*2);
  __half* Hf16  = (__half*)alloc((size_t)BB*EE*NN*2);
  __half* ef16  = (__half*)alloc((size_t)BB*EE*DD*2);
  __half* wW1T  = (__half*)alloc((size_t)HD*2*DD*2);    // (256h x 256d)
  __half* Hw16  = (__half*)alloc((size_t)BB*EE*NN*2);
  __half* etpad = (__half*)alloc((size_t)BB*EE*160*2);
  __half* dW1pT = (__half*)alloc((size_t)HD*160*2);
  __half* fW1pT = (__half*)alloc((size_t)HD*160*2);
  __half* dW2pT = (__half*)alloc((size_t)16*HD*2);
  __half* fW2pT = (__half*)alloc((size_t)16*HD*2);
  __half* hd16  = (__half*)alloc((size_t)BB*EE*HD*2);
  __half* hf16  = (__half*)alloc((size_t)BB*EE*HD*2);
  __half* ete16 = (__half*)alloc((size_t)BB*EE*DD*2);
  __half* EF2T  = (__half*)alloc((size_t)BB*256*EE*2);  // (B, 2D, E)
  __half* Hw2T  = (__half*)alloc((size_t)BB*NN*EE*2);
  __half* cat2  = (__half*)alloc((size_t)BB*NN*512*2);
  __half* qWT   = (__half*)alloc((size_t)DD*384*2);
  __half* kWT   = (__half*)alloc((size_t)DD*DD*2);
  __half* vWT   = (__half*)alloc((size_t)DD*DD*2);
  __half* iqWT  = (__half*)alloc((size_t)DD*DD*2);
  __half* ikWT  = (__half*)alloc((size_t)DD*DD*2);
  __half* ivWT  = (__half*)alloc((size_t)DD*DD*2);
  __half* opWT  = (__half*)alloc((size_t)DD*DD*2);
  __half* oW1T  = (__half*)alloc((size_t)HD*512*2);
  __half* oW2T  = (__half*)alloc((size_t)DD*HD*2);
  __half* Q16   = (__half*)alloc((size_t)BB*NN*DD*2);
  __half* Kp16  = (__half*)alloc((size_t)BB*PP*DD*2);
  __half* Vp16  = (__half*)alloc((size_t)BB*PP*DD*2);
  __half* q216  = (__half*)alloc((size_t)BB*NN*DD*2);
  __half* k216  = (__half*)alloc((size_t)BB*PP*DD*2);
  __half* v216T = (__half*)alloc((size_t)BB*DD*PP*2);   // (B, D, P) transposed
  __half* att16 = (__half*)alloc((size_t)BB*4*NN*PP*2);
  __half* o16   = (__half*)alloc((size_t)BB*NN*DD*2);
  __half* ho16  = (__half*)alloc((size_t)BB*NN*HD*2);
  // f32 buffers
  float* U      = (float*)alloc((size_t)BB*EE*HD*4);
  float* V      = (float*)alloc((size_t)BB*NN*HD*4);
  float* U2     = (float*)alloc((size_t)BB*EE*HD*4);
  float* logits = (float*)alloc((size_t)BB*EE*16*4);
  float* fraw   = (float*)alloc((size_t)BB*EE*16*4);
  float* msg    = (float*)alloc((size_t)BB*NN*256*4);
  float* scores = (float*)alloc((size_t)BB*4*NN*PP*4);
  float* db2p   = (float*)alloc(16*4);
  float* fb2p   = (float*)alloc(16*4);
  float* c0buf  = (float*)alloc(4);

  auto cvt = [&](const float* s, __half* d, int n) {
    k_cvt<<<(n + 255) / 256, 256, 0, stream>>>(s, d, n);
  };
  auto cvtT = [&](const float* s, __half* d, int bat, int rows, int cols) {
    int n = bat * rows * cols;
    k_cvtT<<<(n + 255) / 256, 256, 0, stream>>>(s, d, bat, rows, cols);
  };
  // mode: 0 = f32 out, 1 = f16 out, 2 = f16 out + relu, 3 = f16 out transposed
  auto gemm = [&](int mode, const __half* A, long long sA, int lda,
                  const __half* Bm, long long sB, int ldb,
                  void* C, long long sC, int ldc,
                  const float* bias, float sc, int M, int Nn, int Kd, int bat) {
    GemmP p{A, sA, lda, Bm, sB, ldb, C, sC, ldc, bias, sc, Kd};
    dim3 g(Nn / 16, M / 16, bat);
    switch (mode) {
      case 0: gemm16<0, 0, 0><<<g, 32, 0, stream>>>(p); break;
      case 1: gemm16<1, 0, 0><<<g, 32, 0, stream>>>(p); break;
      case 2: gemm16<1, 1, 0><<<g, 32, 0, stream>>>(p); break;
      default: gemm16<1, 0, 1><<<g, 32, 0, stream>>>(p); break;
    }
  };

  // ---- Phase 0: conversions / padding (weights -> transposed f16) ----
  cvt(af, af16, BB*NN*DD);
  cvtT(af, af16T, BB, NN, DD);
  cvt(pf, pf16, BB*PP*DD);
  cvt(H,  Hf16, BB*EE*NN);
  cvtT(wW1, wW1T, 1, 2*DD, HD);        // (256h rows x 256d cols)
  cvtT(qW, qWT, 1, 384, DD);
  cvtT(kW, kWT, 1, DD, DD);  cvtT(vW, vWT, 1, DD, DD);
  cvtT(iqW, iqWT, 1, DD, DD); cvtT(ikW, ikWT, 1, DD, DD); cvtT(ivW, ivWT, 1, DD, DD);
  cvtT(opW, opWT, 1, DD, DD);
  cvtT(oW1, oW1T, 1, 512, HD); cvtT(oW2, oW2T, 1, HD, DD);
  k_padrowsT<<<(160*HD + 255) / 256, 256, 0, stream>>>(dW1, dW1pT, DD + TT, 160, HD);
  k_padrowsT<<<(160*HD + 255) / 256, 256, 0, stream>>>(fW1, fW1pT, DD + TT, 160, HD);
  k_padcolsT<<<(HD*16 + 255) / 256, 256, 0, stream>>>(dW2, dW2pT, HD, KK, 16);
  k_padcolsT<<<(HD*16 + 255) / 256, 256, 0, stream>>>(fW2, fW2pT, HD, 1, 16);
  k_padbias<<<1, 256, 0, stream>>>(db2, db2p, KK, 16);
  k_padbias<<<1, 256, 0, stream>>>(fb2, fb2p, 1, 16);
  k_c0<<<1, 256, 0, stream>>>(wb1, wW2, wb2, c0buf);

  // ---- Level 1 ----
  // ef_init = H @ af  (f16)  [B^T = af16T (D x N)]
  gemm(1, Hf16, (long long)EE*NN, NN, af16T, (long long)DD*NN, NN,
       ef16, (long long)EE*DD, DD, nullptr, 1.f, EE, DD, NN, BB);
  // U = ef_init @ W1a ; V = af @ W1b + b1   [W1T rows = h, cols = d_global]
  gemm(0, ef16, (long long)EE*DD, DD, wW1T, 0, 2*DD,
       U, (long long)EE*HD, HD, nullptr, 1.f, EE, HD, DD, BB);
  gemm(0, af16, (long long)NN*DD, DD, wW1T + DD, 0, 2*DD,
       V, (long long)NN*HD, HD, wb1, 1.f, NN, HD, DD, BB);
  k_pair1<<<dim3(EE, BB), 256, 0, stream>>>(U, V, wW2, wb2, H, Hw16);
  // edge_features = Hw @ af -> etpad cols 0..127 (f16)
  gemm(1, Hw16, (long long)EE*NN, NN, af16T, (long long)DD*NN, NN,
       etpad, (long long)EE*160, 160, nullptr, 1.f, EE, DD, NN, BB);
  k_etype<<<dim3(EE, BB), 32, 0, stream>>>(H, at, etpad);

  // ---- typed edge embedding ----
  gemm(2, etpad, (long long)EE*160, 160, dW1pT, 0, 160,
       hd16, (long long)EE*HD, HD, db1, 1.f, EE, HD, 160, BB);
  gemm(2, etpad, (long long)EE*160, 160, fW1pT, 0, 160,
       hf16, (long long)EE*HD, HD, fb1, 1.f, EE, HD, 160, BB);
  gemm(0, hd16, (long long)EE*HD, HD, dW2pT, 0, HD,
       logits, (long long)EE*16, 16, db2p, 1.f, EE, 16, HD, BB);
  gemm(0, hf16, (long long)EE*HD, HD, fW2pT, 0, HD,
       fraw, (long long)EE*16, 16, fb2p, 1.f, EE, 16, HD, BB);
  k_ete<<<dim3(EE, BB), 128, 0, stream>>>(logits, fraw, temb, etpad, ete16, EF2T);

  // ---- Level 2 ----
  gemm(0, ete16, (long long)EE*DD, DD, wW1T, 0, 2*DD,
       U2, (long long)EE*HD, HD, nullptr, 1.f, EE, HD, DD, BB);
  k_pair2<<<dim3(EE, BB), 256, 0, stream>>>(U2, V, wW2, wb2, H, c0buf, Hw2T);
  // msg = Hw2^T @ [edge_features | ete]   [B^T = EF2T (256 x E)]
  gemm(0, Hw2T, (long long)NN*EE, EE, EF2T, (long long)256*EE, EE,
       msg, (long long)NN*256, 256, nullptr, 1.f, NN, 256, EE, BB);
  k_cat2a<<<(BB*NN*384 + 255) / 256, 256, 0, stream>>>(msg, af, cat2);

  // ---- cross attention ----
  gemm(1, cat2, (long long)NN*512, 512, qWT, 0, 384,
       Q16, (long long)NN*DD, DD, qb, 1.f, NN, DD, 384, BB);
  gemm(1, pf16, (long long)PP*DD, DD, kWT, 0, DD,
       Kp16, (long long)PP*DD, DD, kb, 1.f, PP, DD, DD, BB);
  gemm(1, pf16, (long long)PP*DD, DD, vWT, 0, DD,
       Vp16, (long long)PP*DD, DD, vb, 1.f, PP, DD, DD, BB);
  gemm(1, Q16, (long long)NN*DD, DD, iqWT, 0, DD,
       q216, (long long)NN*DD, DD, iqb, 1.f, NN, DD, DD, BB);
  gemm(1, Kp16, (long long)PP*DD, DD, ikWT, 0, DD,
       k216, (long long)PP*DD, DD, ikb, 1.f, PP, DD, DD, BB);
  gemm(3, Vp16, (long long)PP*DD, DD, ivWT, 0, DD,      // -> v216T (D x P)
       v216T, (long long)DD*PP, PP, ivb, 1.f, PP, DD, DD, BB);
  const float iscale = 0.17677669529663689f;  // 1/sqrt(32)
  for (int h = 0; h < 4; ++h) {
    // scores_h = q2_h @ k2_h^T : k216 rows are already the "transposed" layout
    gemm(0, q216 + 32 * h, (long long)NN*DD, DD,
         k216 + 32 * h, (long long)PP*DD, DD,
         scores + (long long)h*NN*PP, (long long)4*NN*PP, PP,
         nullptr, iscale, NN, PP, 32, BB);
  }
  k_softrow<<<BB * 4 * NN, 256, 0, stream>>>(scores, att16);
  for (int h = 0; h < 4; ++h) {
    // o_h = att_h @ v_h : B^T = v216T slice (32 x P)
    gemm(1, att16 + (long long)h*NN*PP, (long long)4*NN*PP, PP,
         v216T + (long long)(32 * h) * PP, (long long)DD*PP, PP,
         o16 + 32 * h, (long long)NN*DD, DD, nullptr, 1.f, NN, 32, PP, BB);
  }
  float* pn_out = (float*)d_out + (size_t)BB*NN*DD;  // polyline_new (output 2)
  gemm(0, o16, (long long)NN*DD, DD, opWT, 0, DD,
       pn_out, (long long)NN*DD, DD, opb, 1.f, NN, DD, DD, BB);
  k_cat2b<<<(BB*NN*DD + 255) / 256, 256, 0, stream>>>(pn_out, cat2);

  // ---- output MLP -> agent_new (output 1) ----
  gemm(2, cat2, (long long)NN*512, 512, oW1T, 0, 512,
       ho16, (long long)NN*HD, HD, ob1, 1.f, NN, HD, 512, BB);
  gemm(0, ho16, (long long)NN*HD, HD, oW2T, 0, HD,
       (float*)d_out, (long long)NN*DD, DD, ob2, 1.f, NN, DD, HD, BB);
}
